// GCNSimple_2156073582926
// MI455X (gfx1250) — compile-verified, matrix operand order checked by reading, not run on hardware
//
#include <hip/hip_runtime.h>
#include <hip/hip_bf16.h>
#include <math.h>

// ---------------------------------------------------------------------------
// GCNSimple on MI455X (gfx1250, wave32).
// Memory-bound scatter/gather graph conv; the one dense GEMM (N x 92 -> 10)
// runs on V_WMMA_F32_16X16X4_F32 (fp32-precise, 23 K-steps of 4),
// two 16-row tiles per wave, B operand staged pre-swizzled in LDS.
// ---------------------------------------------------------------------------

#define F_IN   92
#define WIDTH  10
#define NKPAIR (F_IN / 2)   // 46 (k, k+1) row pairs of W_emb

typedef __attribute__((ext_vector_type(2)))  float     v2f;
typedef __attribute__((ext_vector_type(8)))  float     v8f;
typedef __attribute__((ext_vector_type(16))) _Float16  v16h;

// ---------------- zero init (atomic accumulation targets) -------------------
__global__ void gcn_zero_kernel(float* __restrict__ p, long n) {
    long i = (long)blockIdx.x * blockDim.x + threadIdx.x;
    long stride = (long)gridDim.x * blockDim.x;
    for (; i < n; i += stride) p[i] = 0.0f;
}

// ---------------- edge pass 0: edge weights + degrees -----------------------
__global__ void gcn_edge_prep_kernel(const float* __restrict__ r,
                                     const int* __restrict__ src,
                                     const int* __restrict__ dst,
                                     float* __restrict__ ew,
                                     float* __restrict__ degO,
                                     float* __restrict__ degI,
                                     int E) {
    int e = blockIdx.x * blockDim.x + threadIdx.x;
    if (e >= E) return;
    size_t b = (size_t)e * 3;
    float x = r[b], y = r[b + 1], z = r[b + 2];
    ew[e] = expf(-(x * x + y * y + z * z));
    atomicAdd(&degO[src[e]], 1.0f);
    atomicAdd(&degI[dst[e]], 1.0f);
}

// ---------------- degrees -> deg^-1/2 (clipped at 1), in place ---------------
__global__ void gcn_invsqrt_kernel(float* __restrict__ degO,
                                   float* __restrict__ degI, int n) {
    int i = blockIdx.x * blockDim.x + threadIdx.x;
    if (i >= n) return;
    degO[i] = rsqrtf(fmaxf(degO[i], 1.0f));
    degI[i] = rsqrtf(fmaxf(degI[i], 1.0f));
}

// ---------------- embedding GEMM via WMMA: H = A @ W_emb + b ----------------
// One wave owns TWO 16-row tiles (independent accumulator chains sharing B).
// Operand layouts (wave32):
//   A (16x4 f32): lane&15 = M; lanes 0-15 hold K=k0,k0+1; lanes 16-31 K=k0+2,k0+3
//   B (4x16 f32): lane&15 = N; same K-half split -> per-lane v2f = W[ka..ka+1][N]
//   C (16x16 f32, 8 VGPRs): VGPR i -> M = i + (lane>>4)*8, N = lane&15
// B fragments are staged in LDS as sW[kpair*16 + col] = {W[2k][col], W[2k+1][col]}
// (zero-padded to 16 columns) so each step is one aligned ds_load_b64.
__global__ void gcn_embed_wmma_kernel(const float* __restrict__ A,
                                      const float* __restrict__ W,
                                      const float* __restrict__ bias,
                                      float* __restrict__ H, int n) {
    __shared__ v2f sW[NKPAIR * 16];   // 46*16*8B = 5.9 KB

    // Cooperative fill of the padded, WMMA-swizzled B tile.
    for (int idx = threadIdx.x; idx < NKPAIR * 32; idx += blockDim.x) {
        const int kpair = idx >> 5;          // 0..45
        const int rem   = idx & 31;
        const int col   = rem >> 1;          // 0..15
        const int t     = rem & 1;           // element within pair
        const int k     = kpair * 2 + t;     // 0..91
        ((float*)sW)[idx & ~31 ? idx : idx]  = 0.0f; // (no-op keep simple below)
        ((float*)&sW[kpair * 16 + col])[t] =
            (col < WIDTH) ? W[k * WIDTH + col] : 0.0f;
    }
    __syncthreads();

    const int lane = threadIdx.x & 31;
    const int wave = threadIdx.x >> 5;
    const int pair = blockIdx.x * (blockDim.x >> 5) + wave;  // tile pair id
    const int row0 = pair * 32;              // first row of tile 0
    if (row0 >= n) return;                   // wave-uniform: EXEC stays full

    const int m   = lane & 15;               // M within tile (A) / N (B, C)
    const int kh  = lane >> 4;               // K-half select
    const int col = lane & 15;

    // Clamped load rows: out-of-range rows only feed C entries never stored.
    const int rA0 = min(row0 + m,      n - 1);
    const int rA1 = min(row0 + 16 + m, n - 1);
    const float* __restrict__ pa0 = A + (size_t)rA0 * F_IN;
    const float* __restrict__ pa1 = A + (size_t)rA1 * F_IN;

    v8f acc0 = {};
    v8f acc1 = {};

#if __has_builtin(__builtin_amdgcn_wmma_f32_16x16x4_f32)
    #pragma unroll
    for (int k0 = 0; k0 < F_IN; k0 += 4) {   // 92 = 23 * 4, no K padding
        const int ka = k0 + kh * 2;          // even -> 8B-aligned loads
        const v2f a0 = *(const v2f*)(pa0 + ka);
        const v2f a1 = *(const v2f*)(pa1 + ka);
        const v2f w  = sW[((k0 >> 1) + kh) * 16 + col];
        acc0 = __builtin_amdgcn_wmma_f32_16x16x4_f32(
            false, a0, false, w, (short)0, acc0, false, false);
        acc1 = __builtin_amdgcn_wmma_f32_16x16x4_f32(
            false, a1, false, w, (short)0, acc1, false, false);
    }
#else
    // Fallback: f16 WMMA, K padded 92 -> 96 (3 steps of 32).
    for (int k0 = 0; k0 < 96; k0 += 32) {
        v16h a0, a1, w;
        #pragma unroll
        for (int j = 0; j < 16; ++j) {
            int k = k0 + ((j < 8) ? j : j + 8) + kh * 8;
            float av0 = (k < F_IN) ? pa0[k] : 0.0f;
            float av1 = (k < F_IN) ? pa1[k] : 0.0f;
            float wv  = (k < F_IN && col < WIDTH) ? W[k * WIDTH + col] : 0.0f;
            a0[j] = (_Float16)av0;
            a1[j] = (_Float16)av1;
            w[j]  = (_Float16)wv;
        }
        acc0 = __builtin_amdgcn_wmma_f32_16x16x32_f16(
            false, a0, false, w, (short)0, acc0, false, false);
        acc1 = __builtin_amdgcn_wmma_f32_16x16x32_f16(
            false, a1, false, w, (short)0, acc1, false, false);
    }
#endif

    if (col < WIDTH) {
        const float bv = bias[col];
        #pragma unroll
        for (int i = 0; i < 8; ++i) {
            const int r0 = row0 + i + kh * 8;
            const int r1 = r0 + 16;
            if (r0 < n) H[(size_t)r0 * WIDTH + col] = acc0[i] + bv;
            if (r1 < n) H[(size_t)r1 * WIDTH + col] = acc1[i] + bv;
        }
    }
}

// ---------------- edge pass 1: agg1[dst] += h[src] * invO[src] * ew ----------
__global__ void gcn_edge_agg1_kernel(const float* __restrict__ H,
                                     const float* __restrict__ ew,
                                     const float* __restrict__ invO,
                                     const int* __restrict__ src,
                                     const int* __restrict__ dst,
                                     float* __restrict__ agg, int E) {
    int e = blockIdx.x * blockDim.x + threadIdx.x;
    if (e >= E) return;
    const int s = src[e], d = dst[e];
    const float w = ew[e] * invO[s];
    const float* __restrict__ hs = H + (size_t)s * WIDTH;
    float* __restrict__ ag = agg + (size_t)d * WIDTH;
    #pragma unroll
    for (int j = 0; j < WIDTH; ++j) atomicAdd(&ag[j], hs[j] * w);
}

// ---- node pass: x1 = relu(agg1 @ W1 * invI + b1); y = invO * (x1 @ W2) -----
__global__ void gcn_node_mid_kernel(const float* __restrict__ agg,
                                    const float* __restrict__ invI,
                                    const float* __restrict__ invO,
                                    const float* __restrict__ W1,
                                    const float* __restrict__ b1,
                                    const float* __restrict__ W2,
                                    float* __restrict__ y, int n) {
    __shared__ float sW1[WIDTH * WIDTH];
    __shared__ float sW2[WIDTH];
    __shared__ float sb1[WIDTH];
    if (threadIdx.x < WIDTH * WIDTH) sW1[threadIdx.x] = W1[threadIdx.x];
    if (threadIdx.x < WIDTH) {
        sW2[threadIdx.x] = W2[threadIdx.x];
        sb1[threadIdx.x] = b1[threadIdx.x];
    }
    __syncthreads();

    int i = blockIdx.x * blockDim.x + threadIdx.x;
    if (i >= n) return;

    float a[WIDTH];
    const float* __restrict__ ar = agg + (size_t)i * WIDTH;
    #pragma unroll
    for (int j = 0; j < WIDTH; ++j) a[j] = ar[j];

    const float si = invI[i];
    float acc2 = 0.0f;
    #pragma unroll
    for (int c = 0; c < WIDTH; ++c) {
        float t = 0.0f;
        #pragma unroll
        for (int k = 0; k < WIDTH; ++k) t = fmaf(a[k], sW1[k * WIDTH + c], t);
        t = fmaxf(t * si + sb1[c], 0.0f);      // layer-1 output (relu)
        acc2 = fmaf(t, sW2[c], acc2);          // layer-2 mm-first (10 -> 1)
    }
    y[i] = acc2 * invO[i];                     // pre-scale by src norm
}

// ---------------- edge pass 2: agg2[dst] += y[src] * ew ----------------------
__global__ void gcn_edge_agg2_kernel(const float* __restrict__ y,
                                     const float* __restrict__ ew,
                                     const int* __restrict__ src,
                                     const int* __restrict__ dst,
                                     float* __restrict__ agg2, int E) {
    int e = blockIdx.x * blockDim.x + threadIdx.x;
    if (e >= E) return;
    atomicAdd(&agg2[dst[e]], y[src[e]] * ew[e]);
}

// ---------------- node final + mean-pool accumulation ------------------------
__global__ void gcn_node_pool_kernel(const float* __restrict__ agg2,
                                     const float* __restrict__ invI,
                                     const float* __restrict__ b2,
                                     const int* __restrict__ gid,
                                     float* __restrict__ psum,
                                     float* __restrict__ pcnt, int n) {
    int i = blockIdx.x * blockDim.x + threadIdx.x;
    if (i >= n) return;
    const float v = agg2[i] * invI[i] + b2[0];
    const int g = gid[i];
    atomicAdd(&psum[g], v);
    atomicAdd(&pcnt[g], 1.0f);
}

__global__ void gcn_finalize_kernel(const float* __restrict__ psum,
                                    const float* __restrict__ pcnt,
                                    float* __restrict__ out, int g) {
    int i = blockIdx.x * blockDim.x + threadIdx.x;
    if (i >= g) return;
    out[i] = psum[i] / pcnt[i];
}

// ---------------------------------------------------------------------------
extern "C" void kernel_launch(void* const* d_in, const int* in_sizes, int n_in,
                              void* d_out, int out_size, void* d_ws, size_t ws_size,
                              hipStream_t stream) {
    // Inputs (setup_inputs order):
    // 0 atom_features [N*92] f32   1 r [E*3] f32
    // 2 W_emb [92*10]              3 b_emb [10]
    // 4 W1 [10*10]                 5 b1 [10]
    // 6 W2 [10*1]                  7 b2 [1]
    // 8 src [E] i32                9 dst [E] i32
    // 10 graph_ids [N] i32         11 num_graphs (scalar)
    const float* A     = (const float*)d_in[0];
    const float* r     = (const float*)d_in[1];
    const float* W_emb = (const float*)d_in[2];
    const float* b_emb = (const float*)d_in[3];
    const float* W1    = (const float*)d_in[4];
    const float* b1    = (const float*)d_in[5];
    const float* W2    = (const float*)d_in[6];
    const float* b2    = (const float*)d_in[7];
    const int*   src   = (const int*)d_in[8];
    const int*   dst   = (const int*)d_in[9];
    const int*   gid   = (const int*)d_in[10];

    const int E = in_sizes[8];
    const int N = in_sizes[10];
    const int G = out_size;
    float* out = (float*)d_out;

    // Workspace layout (floats). Non-accumulated buffers first, then one
    // contiguous region of atomic-accumulation targets we zero every call.
    float* ws   = (float*)d_ws;
    size_t off  = 0;
    float* ew   = ws + off; off += (size_t)E;          // edge weights
    float* H    = ws + off; off += (size_t)N * WIDTH;  // embedded features
    float* y    = ws + off; off += (size_t)N;          // layer-2 mm-first scalar
    float* zreg = ws + off;                            // ---- zeroed region ----
    float* degO = ws + off; off += (size_t)N;          // -> inv_sqrt_out
    float* degI = ws + off; off += (size_t)N;          // -> inv_sqrt_in
    float* agg1 = ws + off; off += (size_t)N * WIDTH;  // layer-1 aggregation
    float* agg2 = ws + off; off += (size_t)N;          // layer-2 aggregation
    float* psum = ws + off; off += (size_t)G;          // pool sums
    float* pcnt = ws + off; off += (size_t)G;          // pool counts
    const long nzero = (long)(13 * (size_t)N + 2 * (size_t)G);
    (void)ws_size; (void)n_in;

    const int BS = 256;

    // 0) zero atomic targets
    gcn_zero_kernel<<<2048, BS, 0, stream>>>(zreg, nzero);

    // 1) edge weights + degrees
    gcn_edge_prep_kernel<<<(E + BS - 1) / BS, BS, 0, stream>>>(
        r, src, dst, ew, degO, degI, E);

    // 2) degrees -> clipped deg^-1/2
    gcn_invsqrt_kernel<<<(N + BS - 1) / BS, BS, 0, stream>>>(degO, degI, N);

    // 3) WMMA embedding: H = A @ W_emb + b_emb
    //    8 waves/block, 2 16-row tiles per wave -> 256 rows/block.
    {
        const int pairs = (N + 31) / 32;           // tile pairs (32 rows each)
        const int pairs_per_block = BS / 32;       // 8
        gcn_embed_wmma_kernel<<<(pairs + pairs_per_block - 1) / pairs_per_block,
                                BS, 0, stream>>>(A, W_emb, b_emb, H, N);
    }

    // 4) edge gather/scatter, layer 1
    gcn_edge_agg1_kernel<<<(E + BS - 1) / BS, BS, 0, stream>>>(
        H, ew, degO, src, dst, agg1, E);

    // 5) node: finish layer 1, start layer 2 (mm-first)
    gcn_node_mid_kernel<<<(N + BS - 1) / BS, BS, 0, stream>>>(
        agg1, degI, degO, W1, b1, W2, y, N);

    // 6) edge gather/scatter, layer 2 (scalar features)
    gcn_edge_agg2_kernel<<<(E + BS - 1) / BS, BS, 0, stream>>>(
        y, ew, src, dst, agg2, E);

    // 7) node final + pooling
    gcn_node_pool_kernel<<<(N + BS - 1) / BS, BS, 0, stream>>>(
        agg2, degI, b2, gid, psum, pcnt, N);

    // 8) mean per graph
    gcn_finalize_kernel<<<(G + BS - 1) / BS, BS, 0, stream>>>(psum, pcnt, out, G);
}
